// ARC4_65249143160998
// MI455X (gfx1250) — matrix operand order repaired and checked by reading, not compiled, not run on hardware
//
#include <hip/hip_runtime.h>
#include <cstddef>

typedef __attribute__((ext_vector_type(16))) _Float16 v16h;
typedef __attribute__((ext_vector_type(8)))  float    v8f;
typedef __attribute__((ext_vector_type(4)))  unsigned int vu4;

union Frag { vu4 q[2]; v16h h; };

// Branch-free SELU: exp computed unconditionally (single v_exp_f32 TRANS op),
// result picked with v_cndmask instead of EXEC-mask control flow.
__device__ __forceinline__ float selu_f(float x) {
    float e   = __expf(x) - 1.0f;
    float neg = 1.7580993408473766f * e;   // scale*alpha
    float pos = 1.0507009873554805f * x;   // scale
    return (x > 0.0f) ? pos : neg;
}

__device__ __forceinline__ unsigned pack2h(float a, float b) {
    union { _Float16 h[2]; unsigned u; } t;
    t.h[0] = (_Float16)a; t.h[1] = (_Float16)b; return t.u;
}

// Wave-local LDS producer->consumer ordering (no block barrier needed: the
// tile buffers are private per wave; LDS ops are in-order per wave, so only
// the store->cross-lane-load visibility needs a DScnt drain).
__device__ __forceinline__ void wave_lds_fence() {
    asm volatile("s_wait_dscnt 0x0" ::: "memory");
}

// ---------------------------------------------------------------------------
// Weight prep: transpose f32 weights into WMMA B-fragment friendly f16 layout
// Wt[n][k] (n-major, K padded), so lane loads 16 contiguous halves per frag.
// ---------------------------------------------------------------------------
__global__ void gnn_prep_weights(
    const float* __restrict__ W0e, const float* __restrict__ W1e, const float* __restrict__ W2e,
    const float* __restrict__ W0n, const float* __restrict__ W1n, const float* __restrict__ W2n,
    _Float16* __restrict__ Wt0e, _Float16* __restrict__ Wt1e, _Float16* __restrict__ Wt2e,
    _Float16* __restrict__ Wt0n, _Float16* __restrict__ Wt1n, _Float16* __restrict__ Wt2n)
{
    int t = threadIdx.x;
    for (int i = t; i < 64 * 32; i += 256) {          // layer0 -> [64][32]
        int n = i >> 5, k = i & 31;
        Wt0e[i] = (_Float16)((k < 30) ? W0e[k * 64 + n] : 0.0f);
        Wt0n[i] = (_Float16)((k < 20) ? W0n[k * 64 + n] : 0.0f);
    }
    for (int i = t; i < 32 * 64; i += 256) {          // layer1 -> [32][64]
        int n = i >> 6, k = i & 63;
        Wt1e[i] = (_Float16)W1e[k * 32 + n];
        Wt1n[i] = (_Float16)W1n[k * 32 + n];
    }
    for (int i = t; i < 16 * 32; i += 256) {          // layer2 -> [16][32]
        int n = i >> 5, k = i & 31;
        Wt2e[i] = (_Float16)((n < 10) ? W2e[k * 10 + n] : 0.0f);
        Wt2n[i] = (_Float16)((n < 10) ? W2n[k * 10 + n] : 0.0f);
    }
}

// ---------------------------------------------------------------------------
// Encoder: per-row MLP [3 -> 10 (selu) -> 10], f32 in, f16 latent out
// ---------------------------------------------------------------------------
__global__ void gnn_encode(const float* __restrict__ in,
                           const float* __restrict__ W0, const float* __restrict__ b0,
                           const float* __restrict__ W1, const float* __restrict__ b1,
                           _Float16* __restrict__ out, int nRows)
{
    int i = blockIdx.x * blockDim.x + threadIdx.x;
    if (i >= nRows) return;
    float x0 = in[i * 3 + 0], x1 = in[i * 3 + 1], x2 = in[i * 3 + 2];
    float h[10];
#pragma unroll
    for (int j = 0; j < 10; ++j)
        h[j] = selu_f(x0 * W0[j] + x1 * W0[10 + j] + x2 * W0[20 + j] + b0[j]);
#pragma unroll
    for (int j = 0; j < 10; ++j) {
        float v = b1[j];
#pragma unroll
        for (int k = 0; k < 10; ++k) v += h[k] * W1[k * 10 + j];
        out[i * 10 + j] = (_Float16)v;
    }
}

__global__ void gnn_zero(float* __restrict__ p, int n)
{
    for (int i = blockIdx.x * blockDim.x + threadIdx.x; i < n; i += gridDim.x * blockDim.x)
        p[i] = 0.0f;
}

// ---------------------------------------------------------------------------
// Edge block: per 16-edge tile, gather [latE | latN[recv] | latN[send]] (30 f16
// pad 32) and run [32->64 selu ->32 selu ->16] with v_wmma_f32_16x16x32_f16.
// Scatter-add D into agg[recv], accumulate column sums for the global block.
// Each wave works fully independently (private LDS tile, no block barriers).
// ---------------------------------------------------------------------------
#define GNN_WAVES 8

__global__ void __launch_bounds__(256) gnn_edge_block(
    const unsigned* __restrict__ latE, const unsigned* __restrict__ latN,
    const int* __restrict__ senders, const int* __restrict__ receivers,
    const _Float16* __restrict__ Wt0, const float* __restrict__ b0,
    const _Float16* __restrict__ Wt1, const float* __restrict__ b1,
    const _Float16* __restrict__ Wt2, const float* __restrict__ b2,
    _Float16* __restrict__ newE, float* __restrict__ agg, float* __restrict__ esum,
    int nTiles)
{
    __shared__ __align__(16) unsigned smem[GNN_WAVES * 256 + GNN_WAVES * 512];
    __shared__ int srecv[GNN_WAVES * 16];

    const int tid = threadIdx.x;
    const int wave = tid >> 5, lane = tid & 31;
    const int col = lane & 15, kg = lane >> 4;

    unsigned* xb  = smem + wave * 256;                   // 16 rows x 32 halves (x / h2)
    unsigned* h1b = smem + GNN_WAVES * 256 + wave * 512; // 16 rows x 64 halves
    _Float16* h1h = (_Float16*)h1b;
    _Float16* h2h = (_Float16*)xb;
    int* rcvw = srecv + wave * 16;

    // Hoisted, loop-invariant B fragments (per-lane, 2x b128 each).
    Frag B0f[4], B1f[2][2], B2f;
#pragma unroll
    for (int nc = 0; nc < 4; ++nc) {
        const vu4* p = (const vu4*)(Wt0 + ((nc * 16 + col) * 32 + kg * 16));
        B0f[nc].q[0] = p[0]; B0f[nc].q[1] = p[1];
    }
#pragma unroll
    for (int nc = 0; nc < 2; ++nc)
#pragma unroll
        for (int kc = 0; kc < 2; ++kc) {
            const vu4* p = (const vu4*)(Wt1 + ((nc * 16 + col) * 64 + kc * 32 + kg * 16));
            B1f[nc][kc].q[0] = p[0]; B1f[nc][kc].q[1] = p[1];
        }
    {
        const vu4* p = (const vu4*)(Wt2 + (col * 32 + kg * 16));
        B2f.q[0] = p[0]; B2f.q[1] = p[1];
    }
    float b0v[4], b1v[2];
#pragma unroll
    for (int nc = 0; nc < 4; ++nc) b0v[nc] = b0[nc * 16 + col];
#pragma unroll
    for (int nc = 0; nc < 2; ++nc) b1v[nc] = b1[nc * 16 + col];
    float b2v = (col < 10) ? b2[col] : 0.0f;
    float esacc = 0.0f;

    const int gwave  = blockIdx.x * GNN_WAVES + wave;
    const int nWaves = gridDim.x * GNN_WAVES;

    for (int tile = gwave; tile < nTiles; tile += nWaves) {
        // ---- stage x = [latE | latN[recv] | latN[send]] into LDS ----
        if (lane < 16) {
            int e = tile * 16 + lane;
            const unsigned* pe = latE + (size_t)e * 5;
#pragma unroll
            for (int j = 0; j < 5; ++j) xb[lane * 16 + j] = pe[j];
            int r = receivers[e];
            const unsigned* pn = latN + (size_t)r * 5;
#pragma unroll
            for (int j = 0; j < 5; ++j) xb[lane * 16 + 5 + j] = pn[j];
            rcvw[lane] = r;
            xb[lane * 16 + 15] = 0u;   // pad halves 30,31
        } else {
            int m = lane - 16;
            int e = tile * 16 + m;
            int s = senders[e];
            const unsigned* pn = latN + (size_t)s * 5;
#pragma unroll
            for (int j = 0; j < 5; ++j) xb[m * 16 + 10 + j] = pn[j];
        }
        wave_lds_fence();

        // ---- layer 1: K=32, N=64 (4 chunks) ----
        {
            Frag a;
            const vu4* xr = (const vu4*)(xb + col * 16);
            a.q[0] = xr[2 * kg]; a.q[1] = xr[2 * kg + 1];
            v8f cz = {};
#pragma unroll
            for (int nc = 0; nc < 4; ++nc) {
                v8f acc = __builtin_amdgcn_wmma_f32_16x16x32_f16(
                    false, a.h, false, B0f[nc].h, (short)0, cz, false, false);
                float bb = b0v[nc];
#pragma unroll
                for (int r = 0; r < 8; ++r) {
                    float v = selu_f(acc[r] + bb);
                    h1h[(r + kg * 8) * 64 + nc * 16 + col] = (_Float16)v;
                }
            }
        }
        wave_lds_fence();

        // ---- layer 2: K=64 (2 chunks), N=32 (2 chunks) ----
        {
            Frag a0, a1;
            const vu4* hr = (const vu4*)(h1b + col * 32);
            a0.q[0] = hr[2 * kg];     a0.q[1] = hr[2 * kg + 1];
            a1.q[0] = hr[4 + 2 * kg]; a1.q[1] = hr[5 + 2 * kg];
            v8f cz = {};
#pragma unroll
            for (int nc = 0; nc < 2; ++nc) {
                v8f c = __builtin_amdgcn_wmma_f32_16x16x32_f16(
                    false, a0.h, false, B1f[nc][0].h, (short)0, cz, false, false);
                c = __builtin_amdgcn_wmma_f32_16x16x32_f16(
                    false, a1.h, false, B1f[nc][1].h, (short)0, c, false, false);
                float bb = b1v[nc];
#pragma unroll
                for (int r = 0; r < 8; ++r) {
                    float v = selu_f(c[r] + bb);
                    h2h[(r + kg * 8) * 32 + nc * 16 + col] = (_Float16)v;
                }
            }
        }
        wave_lds_fence();

        // ---- layer 3: K=32, N=16 (10 valid) + outputs ----
        {
            Frag a;
            const vu4* xr = (const vu4*)(xb + col * 16);
            a.q[0] = xr[2 * kg]; a.q[1] = xr[2 * kg + 1];
            v8f cz = {};
            v8f o = __builtin_amdgcn_wmma_f32_16x16x32_f16(
                false, a.h, false, B2f.h, (short)0, cz, false, false);
            if (col < 10) {
#pragma unroll
                for (int r = 0; r < 8; ++r) {
                    float v = o[r] + b2v;
                    int m = r + kg * 8;
                    int e = tile * 16 + m;
                    newE[(size_t)e * 10 + col] = (_Float16)v;
                    atomicAdd(agg + (size_t)rcvw[m] * 16 + col, v);
                    esacc += v;
                }
            }
        }
        // LDS ops are in-order per wave: next staging cannot bypass the
        // layer-3 ds_loads above, so no fence needed here.
    }
    if (col < 10) atomicAdd(esum + col, esacc);
}

// ---------------------------------------------------------------------------
// Node block: per 16-node tile, x = [agg(10,f32->f16) | latN(10)] pad 32,
// run [32->64 selu ->32 selu ->16] via WMMA; write f16 latent (+f32 d_out on
// the final round), accumulate column sums for the global block.
// ---------------------------------------------------------------------------
__global__ void __launch_bounds__(256) gnn_node_block(
    const float* __restrict__ agg, const unsigned* __restrict__ latN,
    const _Float16* __restrict__ Wt0, const float* __restrict__ b0,
    const _Float16* __restrict__ Wt1, const float* __restrict__ b1,
    const _Float16* __restrict__ Wt2, const float* __restrict__ b2,
    _Float16* __restrict__ newN, float* __restrict__ nsum, float* __restrict__ outp,
    int nTiles)
{
    __shared__ __align__(16) unsigned smem[GNN_WAVES * 256 + GNN_WAVES * 512];

    const int tid = threadIdx.x;
    const int wave = tid >> 5, lane = tid & 31;
    const int col = lane & 15, kg = lane >> 4;

    unsigned* xb  = smem + wave * 256;
    unsigned* h1b = smem + GNN_WAVES * 256 + wave * 512;
    _Float16* h1h = (_Float16*)h1b;
    _Float16* h2h = (_Float16*)xb;

    Frag B0f[4], B1f[2][2], B2f;
#pragma unroll
    for (int nc = 0; nc < 4; ++nc) {
        const vu4* p = (const vu4*)(Wt0 + ((nc * 16 + col) * 32 + kg * 16));
        B0f[nc].q[0] = p[0]; B0f[nc].q[1] = p[1];
    }
#pragma unroll
    for (int nc = 0; nc < 2; ++nc)
#pragma unroll
        for (int kc = 0; kc < 2; ++kc) {
            const vu4* p = (const vu4*)(Wt1 + ((nc * 16 + col) * 64 + kc * 32 + kg * 16));
            B1f[nc][kc].q[0] = p[0]; B1f[nc][kc].q[1] = p[1];
        }
    {
        const vu4* p = (const vu4*)(Wt2 + (col * 32 + kg * 16));
        B2f.q[0] = p[0]; B2f.q[1] = p[1];
    }
    float b0v[4], b1v[2];
#pragma unroll
    for (int nc = 0; nc < 4; ++nc) b0v[nc] = b0[nc * 16 + col];
#pragma unroll
    for (int nc = 0; nc < 2; ++nc) b1v[nc] = b1[nc * 16 + col];
    float b2v = (col < 10) ? b2[col] : 0.0f;
    float nsacc = 0.0f;

    const int gwave  = blockIdx.x * GNN_WAVES + wave;
    const int nWaves = gridDim.x * GNN_WAVES;

    for (int tile = gwave; tile < nTiles; tile += nWaves) {
        if (lane < 16) {
            int nd = tile * 16 + lane;
            const float* ar = agg + (size_t)nd * 16;
#pragma unroll
            for (int j = 0; j < 5; ++j)
                xb[lane * 16 + j] = pack2h(ar[2 * j], ar[2 * j + 1]);
        } else {
            int m = lane - 16;
            int nd = tile * 16 + m;
            const unsigned* pn = latN + (size_t)nd * 5;
#pragma unroll
            for (int j = 0; j < 5; ++j) xb[m * 16 + 5 + j] = pn[j];
#pragma unroll
            for (int j = 10; j < 16; ++j) xb[m * 16 + j] = 0u;  // pad 20..31
        }
        wave_lds_fence();

        {
            Frag a;
            const vu4* xr = (const vu4*)(xb + col * 16);
            a.q[0] = xr[2 * kg]; a.q[1] = xr[2 * kg + 1];
            v8f cz = {};
#pragma unroll
            for (int nc = 0; nc < 4; ++nc) {
                v8f acc = __builtin_amdgcn_wmma_f32_16x16x32_f16(
                    false, a.h, false, B0f[nc].h, (short)0, cz, false, false);
                float bb = b0v[nc];
#pragma unroll
                for (int r = 0; r < 8; ++r) {
                    float v = selu_f(acc[r] + bb);
                    h1h[(r + kg * 8) * 64 + nc * 16 + col] = (_Float16)v;
                }
            }
        }
        wave_lds_fence();

        {
            Frag a0, a1;
            const vu4* hr = (const vu4*)(h1b + col * 32);
            a0.q[0] = hr[2 * kg];     a0.q[1] = hr[2 * kg + 1];
            a1.q[0] = hr[4 + 2 * kg]; a1.q[1] = hr[5 + 2 * kg];
            v8f cz = {};
#pragma unroll
            for (int nc = 0; nc < 2; ++nc) {
                v8f c = __builtin_amdgcn_wmma_f32_16x16x32_f16(
                    false, a0.h, false, B1f[nc][0].h, (short)0, cz, false, false);
                c = __builtin_amdgcn_wmma_f32_16x16x32_f16(
                    false, a1.h, false, B1f[nc][1].h, (short)0, c, false, false);
                float bb = b1v[nc];
#pragma unroll
                for (int r = 0; r < 8; ++r) {
                    float v = selu_f(c[r] + bb);
                    h2h[(r + kg * 8) * 32 + nc * 16 + col] = (_Float16)v;
                }
            }
        }
        wave_lds_fence();

        {
            Frag a;
            const vu4* xr = (const vu4*)(xb + col * 16);
            a.q[0] = xr[2 * kg]; a.q[1] = xr[2 * kg + 1];
            v8f cz = {};
            v8f o = __builtin_amdgcn_wmma_f32_16x16x32_f16(
                false, a.h, false, B2f.h, (short)0, cz, false, false);
            if (col < 10) {
#pragma unroll
                for (int r = 0; r < 8; ++r) {
                    float v = o[r] + b2v;
                    int m = r + kg * 8;
                    int nd = tile * 16 + m;
                    newN[(size_t)nd * 10 + col] = (_Float16)v;
                    if (outp) outp[(size_t)nd * 10 + col] = v;
                    nsacc += v;
                }
            }
        }
    }
    if (col < 10) atomicAdd(nsum + col, nsacc);
}

// ---------------------------------------------------------------------------
// Global block: single row MLP [20 -> 64 selu -> 32 selu -> 10]
// ---------------------------------------------------------------------------
__global__ void gnn_global_mlp(const float* __restrict__ esum, const float* __restrict__ nsum,
                               const float* __restrict__ W0, const float* __restrict__ b0,
                               const float* __restrict__ W1, const float* __restrict__ b1,
                               const float* __restrict__ W2, const float* __restrict__ b2,
                               float* __restrict__ g)
{
    __shared__ float x[20], h1[64], h2[32];
    int t = threadIdx.x;
    if (t < 10) { x[t] = esum[t]; x[10 + t] = nsum[t]; }
    __syncthreads();
    if (t < 64) {
        float v = b0[t];
        for (int k = 0; k < 20; ++k) v += x[k] * W0[k * 64 + t];
        h1[t] = selu_f(v);
    }
    __syncthreads();
    if (t < 32) {
        float v = b1[t];
        for (int k = 0; k < 64; ++k) v += h1[k] * W1[k * 32 + t];
        h2[t] = selu_f(v);
    }
    __syncthreads();
    if (t < 10) {
        float v = b2[t];
        for (int k = 0; k < 32; ++k) v += h2[k] * W2[k * 10 + t];
        g[t] = v;
    }
}

// ---------------------------------------------------------------------------
extern "C" void kernel_launch(void* const* d_in, const int* in_sizes, int n_in,
                              void* d_out, int out_size, void* d_ws, size_t ws_size,
                              hipStream_t stream)
{
    (void)in_sizes; (void)n_in; (void)out_size; (void)ws_size;
    const int N = 100000, E = 3200000;

    const float* nodes     = (const float*)d_in[0];
    const float* edges     = (const float*)d_in[1];
    const int*   senders   = (const int*)d_in[2];
    const int*   receivers = (const int*)d_in[3];
    // params flattened: dict keys sorted -> edge, global, node, perm_edges, perm_nodes
    const float *eW0 = (const float*)d_in[4],  *eb0 = (const float*)d_in[5];
    const float *eW1 = (const float*)d_in[6],  *eb1 = (const float*)d_in[7];
    const float *eW2 = (const float*)d_in[8],  *eb2 = (const float*)d_in[9];
    const float *gW0 = (const float*)d_in[10], *gb0 = (const float*)d_in[11];
    const float *gW1 = (const float*)d_in[12], *gb1 = (const float*)d_in[13];
    const float *gW2 = (const float*)d_in[14], *gb2 = (const float*)d_in[15];
    const float *nW0 = (const float*)d_in[16], *nb0 = (const float*)d_in[17];
    const float *nW1 = (const float*)d_in[18], *nb1 = (const float*)d_in[19];
    const float *nW2 = (const float*)d_in[20], *nb2 = (const float*)d_in[21];
    const float *peW0 = (const float*)d_in[22], *peb0 = (const float*)d_in[23];
    const float *peW1 = (const float*)d_in[24], *peb1 = (const float*)d_in[25];
    const float *pnW0 = (const float*)d_in[26], *pnb0 = (const float*)d_in[27];
    const float *pnW1 = (const float*)d_in[28], *pnb1 = (const float*)d_in[29];

    char* ws = (char*)d_ws;
    size_t off = 0;
    auto alloc = [&](size_t bytes) -> void* {
        void* p = ws + off;
        off += (bytes + 255) & ~(size_t)255;
        return p;
    };
    unsigned* latE_A = (unsigned*)alloc((size_t)E * 10 * 2);
    unsigned* latE_B = (unsigned*)alloc((size_t)E * 10 * 2);
    unsigned* latN_A = (unsigned*)alloc((size_t)N * 10 * 2);
    unsigned* latN_B = (unsigned*)alloc((size_t)N * 10 * 2);
    float*    agg    = (float*)alloc((size_t)N * 16 * 4 + 128);
    float*    esum   = agg + (size_t)N * 16;
    float*    nsum   = esum + 16;
    _Float16* Wt0e = (_Float16*)alloc(64 * 32 * 2);
    _Float16* Wt1e = (_Float16*)alloc(32 * 64 * 2);
    _Float16* Wt2e = (_Float16*)alloc(16 * 32 * 2);
    _Float16* Wt0n = (_Float16*)alloc(64 * 32 * 2);
    _Float16* Wt1n = (_Float16*)alloc(32 * 64 * 2);
    _Float16* Wt2n = (_Float16*)alloc(16 * 32 * 2);

    gnn_prep_weights<<<1, 256, 0, stream>>>(eW0, eW1, eW2, nW0, nW1, nW2,
                                            Wt0e, Wt1e, Wt2e, Wt0n, Wt1n, Wt2n);

    gnn_encode<<<(N + 255) / 256, 256, 0, stream>>>(nodes, pnW0, pnb0, pnW1, pnb1,
                                                    (_Float16*)latN_A, N);
    gnn_encode<<<(E + 255) / 256, 256, 0, stream>>>(edges, peW0, peb0, peW1, peb1,
                                                    (_Float16*)latE_A, E);

    const int nTilesE = E / 16;             // 200000 (exact)
    const int nTilesN = N / 16;             // 6250   (exact)
    const int eBlocks = 1024;
    const int nBlocks = 512;
    const int zeroN   = N * 16 + 32;

    unsigned *eCur = latE_A, *eNxt = latE_B;
    unsigned *nCur = latN_A, *nNxt = latN_B;

    for (int round = 0; round < 3; ++round) {
        gnn_zero<<<(zeroN + 255) / 256, 256, 0, stream>>>(agg, zeroN);

        gnn_edge_block<<<eBlocks, 256, 0, stream>>>(
            eCur, nCur, senders, receivers,
            Wt0e, eb0, Wt1e, eb1, Wt2e, eb2,
            (_Float16*)eNxt, agg, esum, nTilesE);

        float* outp = (round == 2) ? (float*)d_out : nullptr;
        gnn_node_block<<<nBlocks, 256, 0, stream>>>(
            agg, nCur,
            Wt0n, nb0, Wt1n, nb1, Wt2n, nb2,
            (_Float16*)nNxt, nsum, outp, nTilesN);

        if (round == 2) {
            gnn_global_mlp<<<1, 64, 0, stream>>>(esum, nsum,
                gW0, gb0, gW1, gb1, gW2, gb2, (float*)d_out + (size_t)N * 10);
        }

        unsigned* t;
        t = eCur; eCur = eNxt; eNxt = t;
        t = nCur; nCur = nNxt; nNxt = t;
    }
}
